// SequentialVAE_27315992003201
// MI455X (gfx1250) — matrix operand order repaired and checked by reading, not compiled, not run on hardware
//
#include <hip/hip_runtime.h>
#include <stdint.h>
#include <stddef.h>

typedef _Float16 h16;
typedef __attribute__((ext_vector_type(16))) _Float16 v16h;
typedef __attribute__((ext_vector_type(8)))  _Float16 v8h;
typedef __attribute__((ext_vector_type(8)))  float    v8f;

#define Bb 64
#define Ss 128
#define Tt 127
#define Ll 16
#define Mrows (Tt * Bb)   /* 8128 */

// ---------------------------------------------------------------------------
// WMMA fragment helpers (CDNA5 16x16x32 f16, wave32; A/B in contiguous k order, k = 16*(lane/16) + s)
// A: dense [M,K] row-major (K multiple of 32). Two 16B vector loads per frag.
// B: weights pre-transposed to [N,K]; one 32B vector load per frag.
// ---------------------------------------------------------------------------
__device__ __forceinline__ v16h load_a_frag(const h16* __restrict__ A, int lda,
                                            int row0, int k0) {
  int l  = threadIdx.x & 31;
  int r  = row0 + (l & 15);
  int hi = (l >> 4) & 1;
  const h16* base = A + r * lda + k0 + hi * 16;
  v8h lo  = *(const v8h*)(base);
  v8h hi8 = *(const v8h*)(base + 8);
  return __builtin_shufflevector(lo, hi8, 0, 1, 2, 3, 4, 5, 6, 7, 8, 9, 10,
                                 11, 12, 13, 14, 15);
}

__device__ __forceinline__ v16h load_b_fragT(const h16* __restrict__ Wt, int K,
                                             int k0, int col0) {
  int l  = threadIdx.x & 31;
  int n  = col0 + (l & 15);
  int kh = (l >> 4) & 1;
  return *(const v16h*)(Wt + n * K + k0 + kh * 16);
}

__device__ __forceinline__ void store_tile_f16(h16* dst, int ldc, int tm,
                                               int tn, v8f c) {
  int l   = threadIdx.x & 31;
  int col = tn + (l & 15);
  int rb  = tm + ((l >> 4) & 1) * 8;
#pragma unroll
  for (int r = 0; r < 8; ++r) dst[(rb + r) * ldc + col] = (h16)c[r];
}

__device__ __forceinline__ float sigm(float x) { return 1.f / (1.f + expf(-x)); }

__device__ __forceinline__ uint32_t pcg(uint32_t x) {
  x = x * 747796405u + 2891336453u;
  uint32_t w = ((x >> ((x >> 28u) + 4u)) ^ x) * 277803737u;
  return (w >> 22u) ^ w;
}

// ---------------------------------------------------------------------------
// Generic GEMM: out = act(A[M,K]f16 @ Wt[N,K]f16 + bias). K compile-time.
// One 16x16 C tile per wave.
// ---------------------------------------------------------------------------
template <int K, bool RELU>
__global__ __launch_bounds__(256) void gemm_t(
    const h16* __restrict__ A, const h16* __restrict__ Wt,
    const float* __restrict__ bias, float* __restrict__ outF,
    h16* __restrict__ outH, int M, int N) {
  int wave   = blockIdx.x * 8 + (threadIdx.x >> 5);
  int tilesN = N >> 4;
  int tm     = (wave / tilesN) * 16;
  int tn     = (wave % tilesN) * 16;
  if (tm >= M) return;  // wave-uniform exit
  v8f c = {};
#pragma unroll
  for (int k0 = 0; k0 < K; k0 += 32) {
    v16h a = load_a_frag(A, K, tm, k0);
    v16h b = load_b_fragT(Wt, K, k0, tn);
    c = __builtin_amdgcn_wmma_f32_16x16x32_f16(false, a, false, b, (short)0, c,
                                               false, false);
  }
  int l   = threadIdx.x & 31;
  int col = tn + (l & 15);
  int rb  = tm + ((l >> 4) & 1) * 8;
  float bv = bias[col];
#pragma unroll
  for (int r = 0; r < 8; ++r) {
    float v = c[r] + bv;
    if (RELU) v = fmaxf(v, 0.f);
    int idx = (rb + r) * N + col;
    if (outF) outF[idx] = v;
    if (outH) outH[idx] = (h16)v;
  }
}

// ---------------------------------------------------------------------------
// Tiny feature MLPs: af/rf per (b,t), packed into xs[t*64+b, 0:16 | 16:32] f16
// ---------------------------------------------------------------------------
__global__ __launch_bounds__(256) void encode_feats(
    const float* __restrict__ actions, const float* __restrict__ rewards,
    const float* aw1, const float* ab1, const float* aw2, const float* ab2,
    const float* rw1, const float* rb1, const float* rw2, const float* rb2,
    h16* __restrict__ xs) {
  int i = blockIdx.x * 256 + threadIdx.x;
  if (i >= Bb * Tt) return;
  int b = i / Tt, t = i % Tt;
  float a0 = actions[(b * Ss + t) * 2 + 0];
  float a1 = actions[(b * Ss + t) * 2 + 1];
  float r0 = rewards[b * Ss + t];
  float ha[16], hr[16];
#pragma unroll
  for (int j = 0; j < 16; ++j) {
    ha[j] = fmaxf(a0 * aw1[j] + a1 * aw1[16 + j] + ab1[j], 0.f);
    hr[j] = fmaxf(r0 * rw1[j] + rb1[j], 0.f);
  }
  h16* row = xs + (t * Bb + b) * 32;
#pragma unroll
  for (int k = 0; k < 16; ++k) {
    float va = ab2[k], vr = rb2[k];
#pragma unroll
    for (int j = 0; j < 16; ++j) {
      va += ha[j] * aw2[j * 16 + k];
      vr += hr[j] * rw2[j * 16 + k];
    }
    row[k]      = (h16)va;
    row[16 + k] = (h16)vr;
  }
}

// ---------------------------------------------------------------------------
// Encoder GRU: 1 workgroup, h[64,64] in LDS(f16), per step gh=h@wh via WMMA.
// wh_t transposed [192,64]. gi (with bi) precomputed in global.
// ---------------------------------------------------------------------------
__global__ __launch_bounds__(256) void enc_gru(
    const float* __restrict__ gi,   // [8128,192]
    const h16* __restrict__ wh_t,   // [192,64] transposed
    const float* __restrict__ bh,   // [192]
    h16* __restrict__ enc_hs) {     // [8128,64]
  __shared__ __align__(32) h16 sh[64 * 64];
  __shared__ __align__(32) h16 sgh[64 * 192];
  int tid = threadIdx.x, wv = tid >> 5;
  for (int j = tid; j < 64 * 64; j += 256) sh[j] = (h16)0.f;
  __syncthreads();
  for (int t = 0; t < Tt; ++t) {
#pragma unroll
    for (int i6 = 0; i6 < 6; ++i6) {
      int tle = wv + i6 * 8;
      int tm = (tle / 12) * 16, tn = (tle % 12) * 16;
      v8f c = {};
#pragma unroll
      for (int k0 = 0; k0 < 64; k0 += 32) {
        v16h a = load_a_frag(sh, 64, tm, k0);
        v16h b = load_b_fragT(wh_t, 64, k0, tn);
        c = __builtin_amdgcn_wmma_f32_16x16x32_f16(false, a, false, b,
                                                   (short)0, c, false, false);
      }
      store_tile_f16(sgh, 192, tm, tn, c);
    }
    __syncthreads();
    for (int j = tid; j < 64 * 64; j += 256) {
      int row = j >> 6, col = j & 63;
      const float* g = gi + (t * Bb + row) * 192;
      float ghr = (float)sgh[row * 192 + col]       + bh[col];
      float ghz = (float)sgh[row * 192 + col + 64]  + bh[col + 64];
      float ghn = (float)sgh[row * 192 + col + 128] + bh[col + 128];
      float r  = sigm(g[col] + ghr);
      float z  = sigm(g[col + 64] + ghz);
      float n  = tanhf(g[col + 128] + r * ghn);
      float h  = (float)sh[j];
      float h2 = (1.f - z) * n + z * h;
      sh[j] = (h16)h2;
      enc_hs[(t * Bb + row) * 64 + col] = (h16)h2;
    }
    __syncthreads();
  }
}

// ---------------------------------------------------------------------------
// Reparameterized sampling (deterministic PCG + Box-Muller). Writes f32 latent
// transposed to [B,T,L]; f16 copy with row stride hs (zero-padding if hs=32).
// ---------------------------------------------------------------------------
__global__ __launch_bounds__(256) void sample_k(
    const float* __restrict__ mu, const float* __restrict__ lv,
    float* __restrict__ outF, h16* __restrict__ outH, int hs, uint32_t salt) {
  int i = blockIdx.x * 256 + threadIdx.x;
  if (i >= Mrows * Ll) return;
  uint32_t u1 = pcg((uint32_t)i * 2u + salt);
  uint32_t u2 = pcg((uint32_t)i * 2u + 1u + salt);
  float f1 = (u1 >> 8) * (1.f / 16777216.f) + (0.5f / 16777216.f);
  float f2 = (u2 >> 8) * (1.f / 16777216.f);
  float eps = sqrtf(-2.f * logf(f1)) * cosf(6.28318530718f * f2);
  float v = mu[i] + eps * expf(0.5f * lv[i]);
  int t = i / (Bb * Ll), b = (i / Ll) % Bb, l = i % Ll;
  outF[(b * Tt + t) * Ll + l] = v;
  int row = i / Ll;
  outH[row * hs + l] = (h16)v;
  if (hs == 32) outH[row * hs + 16 + l] = (h16)0.f;  // zero pad K 16->32
}

// llt input, padded to 64 cols: [decfeat(32) | m_t(16) | zeros(16)]
__global__ __launch_bounds__(256) void concat64(
    const h16* __restrict__ a32, const h16* __restrict__ b16,
    h16* __restrict__ out) {
  int i = blockIdx.x * 256 + threadIdx.x;
  if (i >= Mrows * 64) return;
  int row = i >> 6, c = i & 63;
  h16 v = (h16)0.f;
  if (c < 32)      v = a32[row * 32 + c];
  else if (c < 48) v = b16[row * 16 + (c - 32)];
  out[i] = v;
}

// ---------------------------------------------------------------------------
// Fused decoder: each WG owns 64 (t,b) rows. 127 steps of GRU (WMMA gh) +
// BCE head (WMMA 64x64 + dot), masked accumulate -> atomicAdd(d_loss).
// ---------------------------------------------------------------------------
__global__ __launch_bounds__(256) void dec_fused(
    const float* __restrict__ h0,      // [8128,64]
    const float* __restrict__ gi,      // [8128,192]
    const h16* __restrict__ wh_t,      // [192,64] transposed
    const float* __restrict__ bh,      // [192]
    const h16* __restrict__ adw1_t,    // [64,64] transposed
    const float* __restrict__ adb1,    // [64]
    const h16* __restrict__ adw2,      // [64]
    const float* __restrict__ adb2p,   // [1]
    const float* __restrict__ actions, // [64,128,2]
    float* __restrict__ out) {
  __shared__ __align__(32) h16 sh[64 * 64];
  __shared__ __align__(32) h16 sact[64 * 64];
  __shared__ __align__(32) h16 shid[64 * 64];
  __shared__ __align__(32) h16 sgh[64 * 192];
  __shared__ float sred[256];
  int tid = threadIdx.x, wv = tid >> 5;
  int R0 = blockIdx.x * 64;
  for (int j = tid; j < 64 * 64; j += 256) sh[j] = (h16)h0[R0 * 64 + j];
  __syncthreads();
  float adb2 = adb2p[0];
  float acc = 0.f;
  for (int s = 0; s < Tt; ++s) {
    // gh = h @ wh
#pragma unroll
    for (int i6 = 0; i6 < 6; ++i6) {
      int tle = wv + i6 * 8;
      int tm = (tle / 12) * 16, tn = (tle % 12) * 16;
      v8f c = {};
#pragma unroll
      for (int k0 = 0; k0 < 64; k0 += 32) {
        v16h a = load_a_frag(sh, 64, tm, k0);
        v16h b = load_b_fragT(wh_t, 64, k0, tn);
        c = __builtin_amdgcn_wmma_f32_16x16x32_f16(false, a, false, b,
                                                   (short)0, c, false, false);
      }
      store_tile_f16(sgh, 192, tm, tn, c);
    }
    __syncthreads();
    // GRU gates (gi constant across steps)
    for (int j = tid; j < 64 * 64; j += 256) {
      int row = j >> 6, col = j & 63;
      const float* g = gi + (R0 + row) * 192;
      float ghr = (float)sgh[row * 192 + col]       + bh[col];
      float ghz = (float)sgh[row * 192 + col + 64]  + bh[col + 64];
      float ghn = (float)sgh[row * 192 + col + 128] + bh[col + 128];
      float r  = sigm(g[col] + ghr);
      float z  = sigm(g[col + 64] + ghz);
      float n  = tanhf(g[col + 128] + r * ghn);
      float h  = (float)sh[j];
      float h2 = (1.f - z) * n + z * h;
      sh[j]   = (h16)h2;
      sact[j] = (h16)fmaxf(h2, 0.f);
    }
    __syncthreads();
    // hidden = relu(relu(h2) @ adw1 + adb1)
#pragma unroll
    for (int i2 = 0; i2 < 2; ++i2) {
      int tle = wv + i2 * 8;
      int tm = (tle / 4) * 16, tn = (tle % 4) * 16;
      v8f c = {};
#pragma unroll
      for (int k0 = 0; k0 < 64; k0 += 32) {
        v16h a = load_a_frag(sact, 64, tm, k0);
        v16h b = load_b_fragT(adw1_t, 64, k0, tn);
        c = __builtin_amdgcn_wmma_f32_16x16x32_f16(false, a, false, b,
                                                   (short)0, c, false, false);
      }
      int l = tid & 31;
      int col = tn + (l & 15);
      int rb  = tm + ((l >> 4) & 1) * 8;
#pragma unroll
      for (int r = 0; r < 8; ++r)
        shid[(rb + r) * 64 + col] = (h16)fmaxf(c[r] + adb1[col], 0.f);
    }
    __syncthreads();
    // logit + masked BCE
    if (tid < 64) {
      float lg = adb2;
#pragma unroll
      for (int j = 0; j < 64; ++j)
        lg += (float)shid[tid * 64 + j] * (float)adw2[j];
      int R = R0 + tid;
      int t = R >> 6, b = R & 63;
      if (s + t < Tt) {
        int idx = s + t + 1;
        if (idx > Ss - 1) idx = Ss - 1;
        float y  = actions[(b * Ss + idx) * 2 + 1];
        float sp = fmaxf(lg, 0.f) + log1pf(expf(-fabsf(lg)));
        acc += sp - lg * y;
      }
    }
    __syncthreads();
  }
  sred[tid] = acc;
  __syncthreads();
  for (int st = 128; st > 0; st >>= 1) {
    if (tid < st) sred[tid] += sred[tid + st];
    __syncthreads();
  }
  if (tid == 0) atomicAdd(out, sred[0] * (1.f / (float)Bb));
}

// ---------------------------------------------------------------------------
// KL (with the faithful mu/logvar swap in the prior), block-reduced.
// ---------------------------------------------------------------------------
__global__ __launch_bounds__(256) void kl_k(
    const float* __restrict__ mu_m, const float* __restrict__ lv_m,
    const float* __restrict__ mu_t, const float* __restrict__ lv_t,
    float* __restrict__ out) {
  __shared__ float sred[256];
  int i = blockIdx.x * 256 + threadIdx.x;
  float v = 0.f;
  const int NT = Tt * Bb * 2 * Ll;
  if (i < NT) {
    int t = i / (Bb * 2 * Ll);
    int rem = i % (Bb * 2 * Ll);
    int b = rem / (2 * Ll);
    int j = rem % (2 * Ll);
    int o = (t * Bb + b) * Ll;
    float cmu = (j < Ll) ? mu_m[o + j] : mu_t[o + j - Ll];
    float clv = (j < Ll) ? lv_m[o + j] : lv_t[o + j - Ll];
    float pmu = 0.f, plv = 0.f;
    if (t > 0) {
      int po = ((t - 1) * Bb + b) * Ll;
      pmu = (j < Ll) ? lv_m[po + j] : lv_t[po + j - Ll];  // swap (faithful)
      plv = (j < Ll) ? mu_m[po + j] : mu_t[po + j - Ll];  // swap (faithful)
    }
    float d = cmu - pmu;
    v = 0.5f * (plv - clv + (expf(clv) + d * d) * expf(-plv) - 1.f);
  }
  sred[threadIdx.x] = v;
  __syncthreads();
  for (int st = 128; st > 0; st >>= 1) {
    if (threadIdx.x < st) sred[threadIdx.x] += sred[threadIdx.x + st];
    __syncthreads();
  }
  if (threadIdx.x == 0) atomicAdd(out + 1, sred[0] * (1.f / (float)Bb));
}

// Transpose+pad weight [K,N]f32 -> [N,Kpad]f16 (zeros for k>=K).
__global__ __launch_bounds__(256) void cvt_t(const float* __restrict__ s,
                                             h16* __restrict__ d, int K, int N,
                                             int Kpad) {
  int i = blockIdx.x * 256 + threadIdx.x;
  if (i >= N * Kpad) return;
  int n = i / Kpad, k = i % Kpad;
  d[i] = (k < K) ? (h16)s[k * N + n] : (h16)0.f;
}

__global__ void cvt_f16(const float* __restrict__ s, h16* __restrict__ d,
                        int n) {
  int i = blockIdx.x * 256 + threadIdx.x;
  if (i < n) d[i] = (h16)s[i];
}

__global__ void zero2(float* out) {
  out[0] = 0.f;
  out[1] = 0.f;
}

// ---------------------------------------------------------------------------
// Host orchestration
// ---------------------------------------------------------------------------
extern "C" void kernel_launch(void* const* d_in, const int* in_sizes, int n_in,
                              void* d_out, int out_size, void* d_ws,
                              size_t ws_size, hipStream_t stream) {
  const float* actions = (const float*)d_in[0];
  const float* rewards = (const float*)d_in[1];
  const float* AE_W1 = (const float*)d_in[2];  const float* AE_B1 = (const float*)d_in[3];
  const float* AE_W2 = (const float*)d_in[4];  const float* AE_B2 = (const float*)d_in[5];
  const float* RE_W1 = (const float*)d_in[6];  const float* RE_B1 = (const float*)d_in[7];
  const float* RE_W2 = (const float*)d_in[8];  const float* RE_B2 = (const float*)d_in[9];
  const float* MU_W1 = (const float*)d_in[10]; const float* MU_B1 = (const float*)d_in[11];
  const float* MU_W2 = (const float*)d_in[12]; const float* MU_B2 = (const float*)d_in[13];
  const float* LV_W1 = (const float*)d_in[14]; const float* LV_B1 = (const float*)d_in[15];
  const float* LV_W2 = (const float*)d_in[16]; const float* LV_B2 = (const float*)d_in[17];
  const float* EL_W1 = (const float*)d_in[18]; const float* EL_B1 = (const float*)d_in[19];
  const float* EL_W2 = (const float*)d_in[20]; const float* EL_B2 = (const float*)d_in[21];
  const float* MT_W1 = (const float*)d_in[22]; const float* MT_B1 = (const float*)d_in[23];
  const float* MT_W2 = (const float*)d_in[24]; const float* MT_B2 = (const float*)d_in[25];
  const float* LT_W1 = (const float*)d_in[26]; const float* LT_B1 = (const float*)d_in[27];
  const float* LT_W2 = (const float*)d_in[28]; const float* LT_B2 = (const float*)d_in[29];
  const float* LL_W1 = (const float*)d_in[30]; const float* LL_B1 = (const float*)d_in[31];
  const float* LL_W2 = (const float*)d_in[32]; const float* LL_B2 = (const float*)d_in[33];
  const float* LLT_W1 = (const float*)d_in[34]; const float* LLT_B1 = (const float*)d_in[35];
  const float* LLT_W2 = (const float*)d_in[36]; const float* LLT_B2 = (const float*)d_in[37];
  const float* AD_W1 = (const float*)d_in[38]; const float* AD_B1 = (const float*)d_in[39];
  const float* AD_W2 = (const float*)d_in[40]; const float* AD_B2 = (const float*)d_in[41];
  const float* ENC_WI = (const float*)d_in[42]; const float* ENC_WH = (const float*)d_in[43];
  const float* ENC_BI = (const float*)d_in[44]; const float* ENC_BH = (const float*)d_in[45];
  const float* DEC_WI = (const float*)d_in[46]; const float* DEC_WH = (const float*)d_in[47];
  const float* DEC_BI = (const float*)d_in[48]; const float* DEC_BH = (const float*)d_in[49];

  float* out = (float*)d_out;
  char* ws = (char*)d_ws;
  size_t off = 0;
  auto alloc = [&](size_t bytes) -> void* {
    void* p = ws + off;
    off += (bytes + 255) & ~(size_t)255;
    return p;
  };
  // transpose+pad weight [K,N] -> f16 [N,Kpad]
  auto cvtT = [&](const float* src, int K, int N, int Kpad) -> h16* {
    h16* dst = (h16*)alloc((size_t)N * Kpad * sizeof(h16));
    int n = N * Kpad;
    cvt_t<<<(n + 255) / 256, 256, 0, stream>>>(src, dst, K, N, Kpad);
    return dst;
  };

  // f16 transposed weight pool
  h16* enc_wi = cvtT(ENC_WI, 32, 192, 32);
  h16* enc_wh = cvtT(ENC_WH, 64, 192, 64);
  h16* dec_wi = cvtT(DEC_WI, 32, 192, 32);
  h16* dec_wh = cvtT(DEC_WH, 64, 192, 64);
  h16* mu_w1 = cvtT(MU_W1, 64, 64, 64);   h16* mu_w2 = cvtT(MU_W2, 64, 16, 64);
  h16* lv_w1 = cvtT(LV_W1, 64, 64, 64);   h16* lv_w2 = cvtT(LV_W2, 64, 16, 64);
  h16* el_w1 = cvtT(EL_W1, 64, 64, 64);   h16* el_w2 = cvtT(EL_W2, 64, 64, 64);
  h16* mt_w1 = cvtT(MT_W1, 64, 64, 64);   h16* mt_w2 = cvtT(MT_W2, 64, 16, 64);
  h16* lt_w1 = cvtT(LT_W1, 64, 64, 64);   h16* lt_w2 = cvtT(LT_W2, 64, 16, 64);
  h16* ll_w1 = cvtT(LL_W1, 16, 32, 32);   // K padded 16->32
  h16* ll_w2 = cvtT(LL_W2, 32, 32, 32);
  h16* llt_w1 = cvtT(LLT_W1, 48, 64, 64); // K padded 48->64
  h16* llt_w2 = cvtT(LLT_W2, 64, 64, 64);
  h16* ad_w1 = cvtT(AD_W1, 64, 64, 64);
  h16* ad_w2 = (h16*)alloc(64 * sizeof(h16));
  cvt_f16<<<1, 256, 0, stream>>>(AD_W2, ad_w2, 64);

  // activation buffers
  h16*   xs_h    = (h16*)alloc((size_t)Mrows * 32 * 2);
  float* gi_enc  = (float*)alloc((size_t)Mrows * 192 * 4);
  h16*   enc_hs  = (h16*)alloc((size_t)Mrows * 64 * 2);
  h16*   h1      = (h16*)alloc((size_t)Mrows * 64 * 2);
  float* mu_m    = (float*)alloc((size_t)Mrows * 16 * 4);
  float* lv_m    = (float*)alloc((size_t)Mrows * 16 * 4);
  float* mu_t    = (float*)alloc((size_t)Mrows * 16 * 4);
  float* lv_t    = (float*)alloc((size_t)Mrows * 16 * 4);
  h16*   feat    = (h16*)alloc((size_t)Mrows * 64 * 2);
  h16*   m_h     = (h16*)alloc((size_t)Mrows * 32 * 2);  // padded to 32 cols
  h16*   mt_h    = (h16*)alloc((size_t)Mrows * 16 * 2);
  h16*   decfeat = (h16*)alloc((size_t)Mrows * 32 * 2);
  h16*   llt_in  = (h16*)alloc((size_t)Mrows * 64 * 2);  // padded to 64 cols
  float* dec_h0  = (float*)alloc((size_t)Mrows * 64 * 4);
  float* gi_dec  = (float*)alloc((size_t)Mrows * 192 * 4);
  (void)ws_size; (void)n_in; (void)in_sizes; (void)out_size;

  auto g32 = [&](const h16* A, const h16* Wt, const float* bias, float* oF,
                 h16* oH, int N, bool relu) {
    int blocks = ((Mrows / 16) * (N / 16) + 7) / 8;
    if (relu)
      gemm_t<32, true><<<blocks, 256, 0, stream>>>(A, Wt, bias, oF, oH, Mrows, N);
    else
      gemm_t<32, false><<<blocks, 256, 0, stream>>>(A, Wt, bias, oF, oH, Mrows, N);
  };
  auto g64 = [&](const h16* A, const h16* Wt, const float* bias, float* oF,
                 h16* oH, int N, bool relu) {
    int blocks = ((Mrows / 16) * (N / 16) + 7) / 8;
    if (relu)
      gemm_t<64, true><<<blocks, 256, 0, stream>>>(A, Wt, bias, oF, oH, Mrows, N);
    else
      gemm_t<64, false><<<blocks, 256, 0, stream>>>(A, Wt, bias, oF, oH, Mrows, N);
  };

  zero2<<<1, 1, 0, stream>>>(out);

  // feature MLPs -> xs [T*B,32] f16
  encode_feats<<<(Bb * Tt + 255) / 256, 256, 0, stream>>>(
      actions, rewards, AE_W1, AE_B1, AE_W2, AE_B2, RE_W1, RE_B1, RE_W2,
      RE_B2, xs_h);

  // encoder input gates + GRU
  g32(xs_h, enc_wi, ENC_BI, gi_enc, nullptr, 192, false);
  enc_gru<<<1, 256, 0, stream>>>(gi_enc, enc_wh, ENC_BH, enc_hs);

  // mu / lv / el / mut / lvt heads
  g64(enc_hs, mu_w1, MU_B1, nullptr, h1, 64, true);
  g64(h1, mu_w2, MU_B2, mu_m, nullptr, 16, false);
  g64(enc_hs, lv_w1, LV_B1, nullptr, h1, 64, true);
  g64(h1, lv_w2, LV_B2, lv_m, nullptr, 16, false);
  g64(enc_hs, el_w1, EL_B1, nullptr, h1, 64, true);
  g64(h1, el_w2, EL_B2, nullptr, feat, 64, false);
  g64(feat, mt_w1, MT_B1, nullptr, h1, 64, true);
  g64(h1, mt_w2, MT_B2, mu_t, nullptr, 16, false);
  g64(feat, lt_w1, LT_B1, nullptr, h1, 64, true);
  g64(h1, lt_w2, LT_B2, lv_t, nullptr, 16, false);

  // reparameterized samples (also written transposed into d_out)
  int nsmp = Mrows * Ll;
  sample_k<<<(nsmp + 255) / 256, 256, 0, stream>>>(mu_m, lv_m, out + 2, m_h,
                                                   32, 0x9E3779B9u);
  sample_k<<<(nsmp + 255) / 256, 256, 0, stream>>>(
      mu_t, lv_t, out + 2 + Bb * Tt * Ll, mt_h, 16, 0x85EBCA6Bu);

  // decoder feature path
  g32(m_h, ll_w1, LL_B1, nullptr, h1, 32, true);          // h1 as [8128,32]
  g32(h1, ll_w2, LL_B2, nullptr, decfeat, 32, false);
  concat64<<<(Mrows * 64 + 255) / 256, 256, 0, stream>>>(decfeat, mt_h,
                                                         llt_in);
  g64(llt_in, llt_w1, LLT_B1, nullptr, h1, 64, true);
  g64(h1, llt_w2, LLT_B2, dec_h0, nullptr, 64, false);
  g32(decfeat, dec_wi, DEC_BI, gi_dec, nullptr, 192, false);

  // fused decoder scan + BCE loss
  dec_fused<<<Tt, 256, 0, stream>>>(dec_h0, gi_dec, dec_wh, DEC_BH, ad_w1,
                                    AD_B1, ad_w2, AD_B2, actions, out);

  // KL
  int nkl = Tt * Bb * 2 * Ll;
  kl_k<<<(nkl + 255) / 256, 256, 0, stream>>>(mu_m, lv_m, mu_t, lv_t, out);
}
